// RILOOD_29265907155229
// MI455X (gfx1250) — matrix operand order repaired and checked by reading, not compiled, not run on hardware
//
#include <hip/hip_runtime.h>
#include <hip/hip_fp16.h>

#define DEV static __device__ __forceinline__

typedef _Float16 h16_t;
typedef __attribute__((ext_vector_type(8)))  _Float16 v8h;
typedef __attribute__((ext_vector_type(16))) _Float16 v16h;
typedef __attribute__((ext_vector_type(8)))  float    v8f;

// ---------------- problem constants ----------------
#define BATCH  256
#define N1_PER 64
#define N2_PER 32
#define DEG    16
#define HDIM   128
#define LATD   168
#define ENVD   200
#define NLAY   3
#define AFD    31
#define BFD    9
#define NN1    (BATCH * N1_PER)   // 16384
#define NN2    (BATCH * N2_PER)   // 8192
#define EED1   (NN1 * DEG)        // 262144
#define EED2   (NN2 * DEG)        // 131072

DEV v8f vzero8() { v8f z = {0.f,0.f,0.f,0.f,0.f,0.f,0.f,0.f}; return z; }

DEV v8f wmma_f16(v16h a, v16h b, v8f c) {
  // D = A(16x32 f16) * B(32x16 f16) + C(16x16 f32)
  return __builtin_amdgcn_wmma_f32_16x16x32_f16(false, a, false, b, (short)0, c, false, false);
}

// A-matrix fragment (16x32 f16) from a contiguous f16 row of >=kbase+32 elems.
// ISA layout: lanes 0-15 row M=lane, elems 0..7 -> K=kb+0..7, elems 8..15 -> K=kb+16..23
//             lanes 16-31 same rows, elems 0..7 -> K=kb+8..15, elems 8..15 -> K=kb+24..31
DEV v16h a_frag_f16(const h16_t* row, int kbase, int lane) {
  const int o = kbase + ((lane & 16) ? 8 : 0);
  v8h lo = *(const v8h*)(row + o);
  v8h hi = *(const v8h*)(row + o + 16);
  return __builtin_shufflevector(lo, hi, 0,1,2,3,4,5,6,7,8,9,10,11,12,13,14,15);
}

DEV v16h a_frag_f32(const float* row, int kbase, int lane) {
  const int o = kbase + ((lane & 16) ? 8 : 0);
  v16h a;
#pragma unroll
  for (int i = 0; i < 8; ++i) {
    a[i]     = (h16_t)row[o + i];
    a[i + 8] = (h16_t)row[o + 16 + i];
  }
  return a;
}

// ---------------- weight pre-pack for B fragments ----------------
// W: [K][N] row-major f32.  Packed so each (nt,kk) fragment is 32 lanes x 16
// contiguous f16: lane -> col N = nt*16 + (lane&15); elem j -> K = kk*32 + (j&7) + (lane>=16?8:0) + (j>=8?16:0)
__global__ void pack_w_kernel(const float* __restrict__ W, h16_t* __restrict__ P,
                              int K, int Nc) {
  int idx = blockIdx.x * 256 + threadIdx.x;
  if (idx >= K * Nc) return;
  int j    = idx & 15;
  int lane = (idx >> 4) & 31;
  int f    = idx >> 9;
  int KB   = K >> 5;
  int kk   = f % KB;
  int nt   = f / KB;
  int Kidx = kk * 32 + (j & 7) + ((lane & 16) ? 8 : 0) + ((j & 8) ? 16 : 0);
  int Nidx = nt * 16 + (lane & 15);
  P[idx] = (h16_t)W[(size_t)Kidx * Nc + Nidx];
}

// ---------------- embeds ----------------
__global__ void node_embed_kernel(const float* __restrict__ x, const float* __restrict__ W,
                                  const float* __restrict__ b, float* __restrict__ h,
                                  h16_t* __restrict__ hh, int N) {
  int idx = blockIdx.x * 256 + threadIdx.x;
  if (idx >= N * HDIM) return;
  int n = idx >> 7, t = idx & 127;
  const float* xr = x + (size_t)n * AFD;
  float acc = b[t];
#pragma unroll
  for (int a = 0; a < AFD; ++a) acc += xr[a] * W[a * HDIM + t];
  h[idx]  = acc;
  hh[idx] = (h16_t)acc;
}

__global__ void edge_embed_kernel(const float* __restrict__ ea, const float* __restrict__ W,
                                  const float* __restrict__ b, h16_t* __restrict__ ee,
                                  int E) {
  int idx = blockIdx.x * 256 + threadIdx.x;
  if (idx >= E * HDIM) return;
  int e = idx >> 7, t = idx & 127;
  const float* er = ea + (size_t)e * BFD;
  float acc = b[t];
#pragma unroll
  for (int a = 0; a < BFD; ++a) acc += er[a] * W[a * HDIM + t];
  ee[idx] = (h16_t)acc;
}

// ---------------- message MLP: [h_dst | h_src | ee](384) -> relu W1 -> W2, scatter-add ----------------
__global__ __launch_bounds__(128) void msg_mlp_kernel(
    const h16_t* __restrict__ h16, const h16_t* __restrict__ ee16,
    const int* __restrict__ ei, int E,
    const h16_t* __restrict__ W1p, const float* __restrict__ b1,
    const h16_t* __restrict__ W2p, const float* __restrict__ b2,
    float* __restrict__ aggr) {
  __shared__ h16_t lds_hid[4][16 * 136];   // per-wave 16x128 hidden tile (+pad, keeps 16B align)
  const int lane = threadIdx.x & 31;
  const int wave = threadIdx.x >> 5;
  const int tile = blockIdx.x * 4 + wave;  // 16 edges per wave-tile
  const int r    = lane & 15;
  const int half = (lane >> 4) & 1;

  const int* srcI = ei;                    // ei[0] = x_j source
  const int* dstI = ei + E;                // ei[1] = x_i, aggregate target
  const int erow = tile * 16 + r;
  const int src = srcI[erow];
  const int dst = dstI[erow];
  const h16_t* rdst = h16 + (size_t)dst * HDIM;
  const h16_t* rsrc = h16 + (size_t)src * HDIM;
  const h16_t* ree  = ee16 + (size_t)erow * HDIM;

  // GEMM1: [16x384] x [384x128], K = 12 steps of 32, N = 8 tiles of 16
  v8f acc[8];
#pragma unroll
  for (int nt = 0; nt < 8; ++nt) acc[nt] = vzero8();
#pragma unroll
  for (int kk = 0; kk < 12; ++kk) {
    const h16_t* rp = (kk < 4) ? rdst : ((kk < 8) ? rsrc : ree);
    v16h a = a_frag_f16(rp, (kk & 3) * 32, lane);
#pragma unroll
    for (int nt = 0; nt < 8; ++nt) {
      v16h bf = *(const v16h*)(W1p + ((size_t)(nt * 12 + kk) * 32 + lane) * 16);
      acc[nt] = wmma_f16(a, bf, acc[nt]);
    }
  }
  // bias + relu, restripe through LDS (C layout -> A layout)
  h16_t* hid = lds_hid[wave];
#pragma unroll
  for (int nt = 0; nt < 8; ++nt) {
    const int col = nt * 16 + r;
    const float bb = b1[col];
#pragma unroll
    for (int v = 0; v < 8; ++v) {
      const int m = v + half * 8;
      float x = acc[nt][v] + bb;
      hid[m * 136 + col] = (h16_t)(x > 0.f ? x : 0.f);
    }
  }
  __syncthreads();
  // GEMM2: [16x128] x [128x128]
  v8f acc2[8];
#pragma unroll
  for (int nt = 0; nt < 8; ++nt) acc2[nt] = vzero8();
  const h16_t* hrow = hid + r * 136;
#pragma unroll
  for (int kk = 0; kk < 4; ++kk) {
    v16h a = a_frag_f16(hrow, kk * 32, lane);
#pragma unroll
    for (int nt = 0; nt < 8; ++nt) {
      v16h bf = *(const v16h*)(W2p + ((size_t)(nt * 4 + kk) * 32 + lane) * 16);
      acc2[nt] = wmma_f16(a, bf, acc2[nt]);
    }
  }
  int drow[8];
#pragma unroll
  for (int v = 0; v < 8; ++v) drow[v] = dstI[tile * 16 + v + half * 8];
#pragma unroll
  for (int nt = 0; nt < 8; ++nt) {
    const int col = nt * 16 + r;
    const float bb = b2[col];
#pragma unroll
    for (int v = 0; v < 8; ++v)
      unsafeAtomicAdd(&aggr[(size_t)drow[v] * HDIM + col], acc2[nt][v] + bb);
  }
}

// ---------------- update MLP: [h | aggr](256) -> relu W1 -> W2 -> tmp ----------------
__global__ __launch_bounds__(128) void upd_mlp_kernel(
    const h16_t* __restrict__ h16, const float* __restrict__ aggr,
    const h16_t* __restrict__ W1p, const float* __restrict__ b1,
    const h16_t* __restrict__ W2p, const float* __restrict__ b2,
    float* __restrict__ tmp) {
  __shared__ h16_t lds_hid[4][16 * 136];
  const int lane = threadIdx.x & 31;
  const int wave = threadIdx.x >> 5;
  const int tile = blockIdx.x * 4 + wave;  // 16 nodes per wave-tile
  const int r    = lane & 15;
  const int half = (lane >> 4) & 1;
  const int node = tile * 16 + r;
  const h16_t* rh = h16 + (size_t)node * HDIM;
  const float* ra = aggr + (size_t)node * HDIM;

  v8f acc[8];
#pragma unroll
  for (int nt = 0; nt < 8; ++nt) acc[nt] = vzero8();
#pragma unroll
  for (int kk = 0; kk < 8; ++kk) {
    v16h a = (kk < 4) ? a_frag_f16(rh, kk * 32, lane)
                      : a_frag_f32(ra, (kk - 4) * 32, lane);
#pragma unroll
    for (int nt = 0; nt < 8; ++nt) {
      v16h bf = *(const v16h*)(W1p + ((size_t)(nt * 8 + kk) * 32 + lane) * 16);
      acc[nt] = wmma_f16(a, bf, acc[nt]);
    }
  }
  h16_t* hid = lds_hid[wave];
#pragma unroll
  for (int nt = 0; nt < 8; ++nt) {
    const int col = nt * 16 + r;
    const float bb = b1[col];
#pragma unroll
    for (int v = 0; v < 8; ++v) {
      const int m = v + half * 8;
      float x = acc[nt][v] + bb;
      hid[m * 136 + col] = (h16_t)(x > 0.f ? x : 0.f);
    }
  }
  __syncthreads();
  v8f acc2[8];
#pragma unroll
  for (int nt = 0; nt < 8; ++nt) acc2[nt] = vzero8();
  const h16_t* hrow = hid + r * 136;
#pragma unroll
  for (int kk = 0; kk < 4; ++kk) {
    v16h a = a_frag_f16(hrow, kk * 32, lane);
#pragma unroll
    for (int nt = 0; nt < 8; ++nt) {
      v16h bf = *(const v16h*)(W2p + ((size_t)(nt * 4 + kk) * 32 + lane) * 16);
      acc2[nt] = wmma_f16(a, bf, acc2[nt]);
    }
  }
#pragma unroll
  for (int nt = 0; nt < 8; ++nt) {
    const int col = nt * 16 + r;
    const float bb = b2[col];
#pragma unroll
    for (int v = 0; v < 8; ++v) {
      const int m = v + half * 8;
      tmp[(size_t)(tile * 16 + m) * HDIM + col] = acc2[nt][v] + bb;
    }
  }
}

// ---------------- residual + LayerNorm over HDIM, refresh f32 + f16 copies ----------------
__global__ __launch_bounds__(128) void residual_ln_kernel(
    float* __restrict__ h, h16_t* __restrict__ hh, const float* __restrict__ tmp,
    const float* __restrict__ g, const float* __restrict__ bta) {
  const int n = blockIdx.x, t = threadIdx.x;
  __shared__ float s[128];
  __shared__ float st[2];
  float v = h[(size_t)n * HDIM + t] + tmp[(size_t)n * HDIM + t];
  s[t] = v; __syncthreads();
  for (int o = 64; o > 0; o >>= 1) { if (t < o) s[t] += s[t + o]; __syncthreads(); }
  if (t == 0) st[0] = s[0] * (1.f / 128.f);
  __syncthreads();
  float mu = st[0], d = v - mu;
  s[t] = d * d; __syncthreads();
  for (int o = 64; o > 0; o >>= 1) { if (t < o) s[t] += s[t + o]; __syncthreads(); }
  if (t == 0) st[1] = rsqrtf(s[0] * (1.f / 128.f) + 1e-5f);
  __syncthreads();
  float out = d * st[1] * g[t] + bta[t];
  h[(size_t)n * HDIM + t]  = out;
  hh[(size_t)n * HDIM + t] = (h16_t)out;
}

// ---------------- interaction map (one block per graph) ----------------
__global__ __launch_bounds__(256) void interact_kernel(
    const float* __restrict__ h1, const float* __restrict__ h2,
    const float* __restrict__ ln_g, const float* __restrict__ ln_b,
    float* __restrict__ x1e, float* __restrict__ x2e) {
  const int b = blockIdx.x, t = threadIdx.x;
  const float* H1 = h1 + (size_t)b * N1_PER * HDIM;
  const float* H2 = h2 + (size_t)b * N2_PER * HDIM;
  __shared__ float I[N1_PER * N2_PER];
  __shared__ float T[N1_PER * HDIM];
  __shared__ float mu[N1_PER], rs[N1_PER];
  const float scale = 0.08838834764831845f;  // 128^-0.5
  for (int idx = t; idx < N1_PER * N2_PER; idx += 256) {
    int n1 = idx >> 5, n2 = idx & 31;
    float acc = 0.f;
    for (int k = 0; k < HDIM; ++k) acc += H1[n1 * HDIM + k] * H2[n2 * HDIM + k];
    acc *= scale;
    I[idx] = fminf(fmaxf(acc, -10.f), 10.f);
  }
  __syncthreads();
  // h1i = LN(I @ H2)
  for (int idx = t; idx < N1_PER * HDIM; idx += 256) {
    int n1 = idx >> 7, c = idx & 127;
    float acc = 0.f;
    for (int m = 0; m < N2_PER; ++m) acc += I[n1 * N2_PER + m] * H2[m * HDIM + c];
    T[idx] = acc;
  }
  __syncthreads();
  if (t < N1_PER) {
    float s = 0.f;
    for (int c = 0; c < HDIM; ++c) s += T[t * HDIM + c];
    float m = s * (1.f / HDIM), vv = 0.f;
    for (int c = 0; c < HDIM; ++c) { float d = T[t * HDIM + c] - m; vv += d * d; }
    mu[t] = m; rs[t] = rsqrtf(vv * (1.f / HDIM) + 1e-5f);
  }
  __syncthreads();
  for (int idx = t; idx < N1_PER * HDIM; idx += 256) {
    int n1 = idx >> 7, c = idx & 127;
    float* o = x1e + ((size_t)b * N1_PER + n1) * (2 * HDIM);
    o[c]        = H1[n1 * HDIM + c];
    o[HDIM + c] = (T[idx] - mu[n1]) * rs[n1] * ln_g[c] + ln_b[c];
  }
  __syncthreads();
  // h2i = LN(I^T @ H1)
  for (int idx = t; idx < N2_PER * HDIM; idx += 256) {
    int n2 = idx >> 7, c = idx & 127;
    float acc = 0.f;
    for (int m = 0; m < N1_PER; ++m) acc += I[m * N2_PER + n2] * H1[m * HDIM + c];
    T[idx] = acc;
  }
  __syncthreads();
  if (t < N2_PER) {
    float s = 0.f;
    for (int c = 0; c < HDIM; ++c) s += T[t * HDIM + c];
    float m = s * (1.f / HDIM), vv = 0.f;
    for (int c = 0; c < HDIM; ++c) { float d = T[t * HDIM + c] - m; vv += d * d; }
    mu[t] = m; rs[t] = rsqrtf(vv * (1.f / HDIM) + 1e-5f);
  }
  __syncthreads();
  for (int idx = t; idx < N2_PER * HDIM; idx += 256) {
    int n2 = idx >> 7, c = idx & 127;
    float* o = x2e + ((size_t)b * N2_PER + n2) * (2 * HDIM);
    o[c]        = H2[n2 * HDIM + c];
    o[HDIM + c] = (T[idx] - mu[n2]) * rs[n2] * ln_g[c] + ln_b[c];
  }
}

// ---------------- Set2Set (3 steps) + fingerprint projection ----------------
DEV float sigmf(float x) { return 1.f / (1.f + expf(-x)); }

__global__ __launch_bounds__(256) void set2set_kernel(
    const float* __restrict__ xe, int n,
    const float* __restrict__ Wih, const float* __restrict__ Whh,
    const float* __restrict__ bih, const float* __restrict__ bhh,
    const float* __restrict__ fpW, const float* __restrict__ fpb,
    float* __restrict__ gout) {
  const int b = blockIdx.x, t = threadIdx.x;
  const float* X = xe + (size_t)b * n * 256;
  __shared__ float hs[256], cs[256], q[512], gates[1024], e[64], r[256];
  hs[t] = 0.f; cs[t] = 0.f; q[t] = 0.f; q[t + 256] = 0.f;
  __syncthreads();
  for (int step = 0; step < 3; ++step) {
    for (int o = t; o < 1024; o += 256) {
      float acc = bih[o] + bhh[o];
      const float* wr = Wih + (size_t)o * 512;
      for (int k = 0; k < 512; ++k) acc += wr[k] * q[k];
      const float* hr = Whh + (size_t)o * 256;
      for (int k = 0; k < 256; ++k) acc += hr[k] * hs[k];
      gates[o] = acc;
    }
    __syncthreads();
    {
      float ig = sigmf(gates[t]),     fg = sigmf(gates[256 + t]);
      float gg = tanhf(gates[512 + t]), og = sigmf(gates[768 + t]);
      float c  = fg * cs[t] + ig * gg;
      cs[t] = c; hs[t] = og * tanhf(c);
    }
    __syncthreads();
    if (t < n) {
      float acc = 0.f;
      for (int k = 0; k < 256; ++k) acc += X[t * 256 + k] * hs[k];
      e[t] = acc;
    }
    __syncthreads();
    if (t == 0) {
      float mx = -1e30f;
      for (int m = 0; m < n; ++m) mx = fmaxf(mx, e[m]);
      float sm = 0.f;
      for (int m = 0; m < n; ++m) { e[m] = expf(e[m] - mx); sm += e[m]; }
      float inv = 1.f / sm;
      for (int m = 0; m < n; ++m) e[m] *= inv;
    }
    __syncthreads();
    {
      float acc = 0.f;
      for (int m = 0; m < n; ++m) acc += e[m] * X[m * 256 + t];
      r[t] = acc;
    }
    __syncthreads();
    q[t] = hs[t]; q[256 + t] = r[t];
    __syncthreads();
  }
  if (t < HDIM) {
    float acc = fpb[t];
    for (int k = 0; k < 512; ++k) acc += q[k] * fpW[k * HDIM + t];
    gout[(size_t)b * HDIM + t] = acc;
  }
}

// ---------------- MCVAE encode + reparam + MCAR fusion + predictor ----------------
__global__ __launch_bounds__(256) void head_kernel(
    const float* __restrict__ g1, const float* __restrict__ g2,
    const int* __restrict__ env_idx, const float* __restrict__ eps,
    const float* encW1, const float* encb1, const float* encW2, const float* encb2,
    const float* gW1, const float* gb1, const float* gW2, const float* gb2,
    const float* lW, const float* lb, const float* prelu_a,
    const float* oW, const float* ob, const float* mcg, const float* mcb,
    const float* pW1, const float* pb1, const float* pW2, const float* pb2,
    float* __restrict__ out) {
  const int b = blockIdx.x, t = threadIdx.x;
  __shared__ float t1[128], enc[336], z[168], Ev[296], Oc1[168], Hc[168], hcv[168], p1[84], st[2];
  const float* G1 = g1 + (size_t)b * HDIM;
  int env = env_idx[b]; env = env < 0 ? 0 : (env > ENVD - 1 ? ENVD - 1 : env);
  if (t < 128) {
    float acc = encb1[t];
    for (int k = 0; k < 128; ++k) acc += G1[k] * encW1[k * 128 + t];
    acc += encW1[(128 + env) * 128 + t];   // one-hot pick
    t1[t] = fmaxf(acc, 0.f);
  }
  __syncthreads();
  for (int o = t; o < 336; o += 256) {
    float acc = encb2[o];
    for (int k = 0; k < 128; ++k) acc += t1[k] * encW2[k * 336 + o];
    enc[o] = acc;
  }
  __syncthreads();
  if (t < 168) {
    float mu = fminf(fmaxf(enc[t], -10.f), 10.f);
    float lv = fminf(fmaxf(enc[168 + t], -10.f), 10.f);
    z[t]  = mu + eps[(size_t)b * 168 + t] * expf(0.5f * lv);
    Ev[t] = z[t];
  }
  if (t < 128) Ev[168 + t] = g2[(size_t)b * HDIM + t];
  __syncthreads();
  if (t < 168) {
    float acc = gb1[t];
    for (int k = 0; k < 296; ++k) acc += Ev[k] * gW1[k * 168 + t];
    Oc1[t] = fmaxf(acc, 0.f);
  }
  __syncthreads();
  if (t < 168) {
    float oc = gb2[t];
    for (int k = 0; k < 168; ++k) oc += Oc1[k] * gW2[k * 168 + t];
    float pre = lb[t];
    for (int k = 0; k < 296; ++k) pre += Ev[k] * lW[k * 168 + t];
    float of = pre >= 0.f ? pre : prelu_a[0] * pre;
    Hc[t] = oc * of;
  }
  __syncthreads();
  if (t < 168) {
    float acc = ob[t];
    for (int k = 0; k < 168; ++k) acc += Hc[k] * oW[k * 168 + t];
    hcv[t] = acc + z[t];
  }
  __syncthreads();
  if (t == 0) {
    float s = 0.f;
    for (int k = 0; k < 168; ++k) s += hcv[k];
    float m = s / 168.f, vv = 0.f;
    for (int k = 0; k < 168; ++k) { float d = hcv[k] - m; vv += d * d; }
    st[0] = m; st[1] = rsqrtf(vv / 168.f + 1e-5f);
  }
  __syncthreads();
  if (t < 168) hcv[t] = (hcv[t] - st[0]) * st[1] * mcg[t] + mcb[t];
  __syncthreads();
  if (t < 84) {
    float acc = pb1[t];
    for (int k = 0; k < 168; ++k) acc += hcv[k] * pW1[k * 84 + t];
    p1[t] = fmaxf(acc, 0.f);
  }
  __syncthreads();
  if (t == 0) {
    float acc = pb2[0];
    for (int k = 0; k < 84; ++k) acc += p1[k] * pW2[k];
    out[b] = acc;
  }
}

// ---------------- host orchestration ----------------
extern "C" void kernel_launch(void* const* d_in, const int* in_sizes, int n_in,
                              void* d_out, int out_size, void* d_ws, size_t ws_size,
                              hipStream_t stream) {
  (void)in_sizes; (void)n_in; (void)out_size; (void)ws_size;
  const float* x1  = (const float*)d_in[0];
  const int*   ei1 = (const int*)  d_in[1];
  const float* ea1 = (const float*)d_in[2];
  const float* x2  = (const float*)d_in[3];
  const int*   ei2 = (const int*)  d_in[4];
  const float* ea2 = (const float*)d_in[5];
  const int*   env = (const int*)  d_in[6];
  const float* eps = (const float*)d_in[7];

  // encoder param blocks (setup_inputs insertion order, 20 tensors each)
  enum { NODE_W, NODE_B, EDGE_W, EDGE_B, MSG_W1, MSG_B1, MSG_W2, MSG_B2,
         UPD_W1, UPD_B1, UPD_W2, UPD_B2, LN_G, LN_B,
         LSTM_WIH, LSTM_WHH, LSTM_BIH, LSTM_BHH, FP_W, FP_B };
  struct EncP { const float* p[20]; };
  EncP encs[2];
  for (int e = 0; e < 2; ++e)
    for (int i = 0; i < 20; ++i)
      encs[e].p[i] = (const float*)d_in[8 + 20 * e + i];

  int pi = 48;
  const float* im_ln_g = (const float*)d_in[pi++];
  const float* im_ln_b = (const float*)d_in[pi++];
  const float* enc_W1  = (const float*)d_in[pi++];
  const float* enc_b1  = (const float*)d_in[pi++];
  const float* enc_W2  = (const float*)d_in[pi++];
  const float* enc_b2  = (const float*)d_in[pi++];
  const float* g_W1    = (const float*)d_in[pi++];
  const float* g_b1    = (const float*)d_in[pi++];
  const float* g_W2    = (const float*)d_in[pi++];
  const float* g_b2    = (const float*)d_in[pi++];
  const float* l_W     = (const float*)d_in[pi++];
  const float* l_b     = (const float*)d_in[pi++];
  const float* prelu_a = (const float*)d_in[pi++];
  const float* o_W     = (const float*)d_in[pi++];
  const float* o_b     = (const float*)d_in[pi++];
  const float* mc_ln_g = (const float*)d_in[pi++];
  const float* mc_ln_b = (const float*)d_in[pi++];
  const float* p_W1    = (const float*)d_in[pi++];
  const float* p_b1    = (const float*)d_in[pi++];
  const float* p_W2    = (const float*)d_in[pi++];
  const float* p_b2    = (const float*)d_in[pi++];

  // ---- workspace carve ----
  char* w = (char*)d_ws; size_t off = 0;
  auto take = [&](size_t n) { char* p = w + off; off += (n + 255) & ~(size_t)255; return p; };
  float* h1   = (float*)take((size_t)NN1 * HDIM * 4);
  h16_t* h1h  = (h16_t*)take((size_t)NN1 * HDIM * 2);
  float* h2   = (float*)take((size_t)NN2 * HDIM * 4);
  h16_t* h2h  = (h16_t*)take((size_t)NN2 * HDIM * 2);
  h16_t* ee1h = (h16_t*)take((size_t)EED1 * HDIM * 2);
  h16_t* ee2h = (h16_t*)take((size_t)EED2 * HDIM * 2);
  float* aggr = (float*)take((size_t)NN1 * HDIM * 4);
  float* tmp  = (float*)take((size_t)NN1 * HDIM * 4);
  float* x1e  = (float*)take((size_t)BATCH * N1_PER * 2 * HDIM * 4);
  float* x2e  = (float*)take((size_t)BATCH * N2_PER * 2 * HDIM * 4);
  float* g1   = (float*)take((size_t)BATCH * HDIM * 4);
  float* g2   = (float*)take((size_t)BATCH * HDIM * 4);
  h16_t *mW1p[2], *mW2p[2], *uW1p[2], *uW2p[2];
  for (int e = 0; e < 2; ++e) {
    mW1p[e] = (h16_t*)take((size_t)NLAY * 384 * HDIM * 2);
    mW2p[e] = (h16_t*)take((size_t)NLAY * HDIM * HDIM * 2);
    uW1p[e] = (h16_t*)take((size_t)NLAY * 256 * HDIM * 2);
    uW2p[e] = (h16_t*)take((size_t)NLAY * HDIM * HDIM * 2);
  }

  // ---- pack weights into WMMA B-fragment layout ----
  for (int e = 0; e < 2; ++e)
    for (int l = 0; l < NLAY; ++l) {
      pack_w_kernel<<<(384 * 128) / 256, 256, 0, stream>>>(encs[e].p[MSG_W1] + (size_t)l * 384 * 128, mW1p[e] + (size_t)l * 384 * 128, 384, 128);
      pack_w_kernel<<<(128 * 128) / 256, 256, 0, stream>>>(encs[e].p[MSG_W2] + (size_t)l * 128 * 128, mW2p[e] + (size_t)l * 128 * 128, 128, 128);
      pack_w_kernel<<<(256 * 128) / 256, 256, 0, stream>>>(encs[e].p[UPD_W1] + (size_t)l * 256 * 128, uW1p[e] + (size_t)l * 256 * 128, 256, 128);
      pack_w_kernel<<<(128 * 128) / 256, 256, 0, stream>>>(encs[e].p[UPD_W2] + (size_t)l * 128 * 128, uW2p[e] + (size_t)l * 128 * 128, 128, 128);
    }

  // ---- embeds ----
  node_embed_kernel<<<NN1 * HDIM / 256, 256, 0, stream>>>(x1, encs[0].p[NODE_W], encs[0].p[NODE_B], h1, h1h, NN1);
  node_embed_kernel<<<NN2 * HDIM / 256, 256, 0, stream>>>(x2, encs[1].p[NODE_W], encs[1].p[NODE_B], h2, h2h, NN2);
  edge_embed_kernel<<<EED1 * HDIM / 256, 256, 0, stream>>>(ea1, encs[0].p[EDGE_W], encs[0].p[EDGE_B], ee1h, EED1);
  edge_embed_kernel<<<EED2 * HDIM / 256, 256, 0, stream>>>(ea2, encs[1].p[EDGE_W], encs[1].p[EDGE_B], ee2h, EED2);

  // ---- MPNN layers ----
  for (int e = 0; e < 2; ++e) {
    const int* ei = e ? ei2 : ei1;
    const int  E  = e ? EED2 : EED1;
    const int  N  = e ? NN2 : NN1;
    float* h   = e ? h2 : h1;
    h16_t* hh  = e ? h2h : h1h;
    h16_t* eh  = e ? ee2h : ee1h;
    for (int l = 0; l < NLAY; ++l) {
      hipMemsetAsync(aggr, 0, (size_t)N * HDIM * 4, stream);
      msg_mlp_kernel<<<E / 64, 128, 0, stream>>>(hh, eh, ei, E,
          mW1p[e] + (size_t)l * 384 * HDIM, encs[e].p[MSG_B1] + (size_t)l * HDIM,
          mW2p[e] + (size_t)l * HDIM * HDIM, encs[e].p[MSG_B2] + (size_t)l * HDIM, aggr);
      upd_mlp_kernel<<<N / 64, 128, 0, stream>>>(hh, aggr,
          uW1p[e] + (size_t)l * 256 * HDIM, encs[e].p[UPD_B1] + (size_t)l * HDIM,
          uW2p[e] + (size_t)l * HDIM * HDIM, encs[e].p[UPD_B2] + (size_t)l * HDIM, tmp);
      residual_ln_kernel<<<N, 128, 0, stream>>>(h, hh, tmp,
          encs[e].p[LN_G] + (size_t)l * HDIM, encs[e].p[LN_B] + (size_t)l * HDIM);
    }
  }

  // ---- interaction map + Set2Set + head ----
  interact_kernel<<<BATCH, 256, 0, stream>>>(h1, h2, im_ln_g, im_ln_b, x1e, x2e);
  set2set_kernel<<<BATCH, 256, 0, stream>>>(x1e, N1_PER,
      encs[0].p[LSTM_WIH], encs[0].p[LSTM_WHH], encs[0].p[LSTM_BIH], encs[0].p[LSTM_BHH],
      encs[0].p[FP_W], encs[0].p[FP_B], g1);
  set2set_kernel<<<BATCH, 256, 0, stream>>>(x2e, N2_PER,
      encs[1].p[LSTM_WIH], encs[1].p[LSTM_WHH], encs[1].p[LSTM_BIH], encs[1].p[LSTM_BHH],
      encs[1].p[FP_W], encs[1].p[FP_B], g2);
  head_kernel<<<BATCH, 256, 0, stream>>>(g1, g2, env, eps,
      enc_W1, enc_b1, enc_W2, enc_b2, g_W1, g_b1, g_W2, g_b2,
      l_W, l_b, prelu_a, o_W, o_b, mc_ln_g, mc_ln_b,
      p_W1, p_b1, p_W2, p_b2, (float*)d_out);
}
